// SemanticAwareSuperLayer_9792525435095
// MI455X (gfx1250) — matrix operand order repaired and checked by reading, not compiled, not run on hardware
//
#include <hip/hip_runtime.h>
#include <math.h>
#include <stdint.h>

typedef __bf16 v16bf __attribute__((ext_vector_type(16)));
typedef float  v8f   __attribute__((ext_vector_type(8)));

#define D_MODEL 1024
#define D_FF    4096
#define RANK    128
#define M_TOTAL 8192
#define MT      32      // rows per workgroup (2 half-tiles of 16)
#define MTH     16
#define NTHREADS 256    // 8 waves: 2 M-halves x 4 N/F-slices
#define NWAVES  8

// ---------------------------------------------------------------------------
// Workspace: bf16 hi/lo weight-fragment buffers, pre-swizzled into the exact
// WMMA B-operand VGPR layout (lane-major, 8 dwords per lane per fragment).
// Offsets in uint16_t units. Total 5.24 MB -> L2-resident (192 MB L2).
// ---------------------------------------------------------------------------
#define SZ_B1 (D_MODEL * RANK)      // Us1  = cfc_U*S1   K=1024 N=128
#define SZ_B2 (RANK * D_FF)         // V1^T              K=128  N=4096
#define SZ_B3 (D_FF * RANK)         // Us2  = cproj_U*S2 K=4096 N=128
#define SZ_B4 (RANK * D_MODEL)      // V2^T              K=128  N=1024
#define B1H 0
#define B1L (B1H + SZ_B1)
#define B2H (B1L + SZ_B1)
#define B2L (B2H + SZ_B2)
#define B3H (B2L + SZ_B2)
#define B3L (B3H + SZ_B3)
#define B4H (B3L + SZ_B3)
#define B4L (B4H + SZ_B4)

// hi/lo split for bf16x3 compensated GEMM (truncation; lo captures residual)
__device__ __forceinline__ void split_bf(float x, uint16_t& h, uint16_t& l) {
  uint32_t b  = __float_as_uint(x);
  uint32_t ht = b & 0xffff0000u;
  h = (uint16_t)(ht >> 16);
  float r = x - __uint_as_float(ht);
  l = (uint16_t)(__float_as_uint(r) >> 16);
}

// B-operand (32x16, 16-bit) fragment element position (u16 index)
__device__ __forceinline__ size_t bpos(int N, int k, int n) {
  int chunk = k >> 5, kin = k & 31;
  int lane  = (n & 15) | (((kin >> 3) & 1) << 4);
  int v     = ((kin & 16) >> 2) + ((kin & 7) >> 1);
  return ((((size_t)chunk * (N >> 4) + (n >> 4)) * 32 + lane) * 8 + v) * 2 + (kin & 1);
}

// A-operand (16x32, 16-bit) fragment element position within one K-chunk (u16 index)
__device__ __forceinline__ int apos(int chunk, int m, int kin) {
  int lane = m | (((kin >> 3) & 1) << 4);
  int v    = ((kin & 16) >> 2) + ((kin & 7) >> 1);
  return ((chunk * 32 + lane) * 8 + v) * 2 + (kin & 1);
}

__device__ __forceinline__ void store_a16(uint16_t* hi, uint16_t* lo,
                                          int chunk, int m, int kin, float x) {
  uint16_t h, l; split_bf(x, h, l);
  int p = apos(chunk, m, kin);
  hi[p] = h; lo[p] = l;
}

union FragU { uint4 q[2]; v16bf v; };

__device__ __forceinline__ v16bf load_frag(const uint16_t* base, int frag, int lane) {
  const uint4* p = (const uint4*)(base + ((size_t)frag * 32 + lane) * 16);
  FragU u; u.q[0] = p[0]; u.q[1] = p[1];
  return u.v;
}

// bf16x3: acc += hi*hi + hi*lo + lo*hi  (lo*lo term ~2^-16, dropped)
__device__ __forceinline__ v8f mma3(v8f acc, v16bf ah, v16bf al, v16bf bh, v16bf bl) {
  acc = __builtin_amdgcn_wmma_f32_16x16x32_bf16(false, ah, false, bh, (short)0, acc, false, false);
  acc = __builtin_amdgcn_wmma_f32_16x16x32_bf16(false, ah, false, bl, (short)0, acc, false, false);
  acc = __builtin_amdgcn_wmma_f32_16x16x32_bf16(false, al, false, bh, (short)0, acc, false, false);
  return acc;
}

// Fast erf (A&S 7.1.26, abs err ~1.5e-7): ~10 FMAs + v_rcp_f32 + v_exp_f32.
// rcp/exp run on the TRANS pipe and co-execute with the XDL WMMA stream.
__device__ __forceinline__ float erf_fast(float x) {
  float ax = fabsf(x);
  float t  = __builtin_amdgcn_rcpf(fmaf(0.3275911f, ax, 1.0f));
  float p  = t * fmaf(t, fmaf(t, fmaf(t, fmaf(t, 1.061405429f, -1.453152027f),
                                      1.421413741f), -0.284496736f), 0.254829592f);
  float r  = fmaf(-p, __expf(-ax * ax), 1.0f);
  return copysignf(r, x);
}

__device__ __forceinline__ float gelu_exact(float x) {
  return 0.5f * x * (1.0f + erf_fast(x * 0.70710678118654752f));
}

// ---------------------------------------------------------------------------
// Prologue: fold S into U factors, transpose V factors, split hi/lo, and write
// straight into WMMA B-fragment order so the main kernel's weight loads are
// pure coalesced global_load_b128.
// ---------------------------------------------------------------------------
__global__ void prep_weights(const float* __restrict__ cfc_U,
                             const float* __restrict__ cfc_S,
                             const float* __restrict__ cfc_V,
                             const float* __restrict__ cproj_U,
                             const float* __restrict__ cproj_S,
                             const float* __restrict__ cproj_V,
                             uint16_t* __restrict__ ws) {
  int i = blockIdx.x * blockDim.x + threadIdx.x;
  if (i < SZ_B1) {                        // Us1: k=d, n=r
    int k = i >> 7, n = i & 127;
    float w = cfc_U[i] * cfc_S[n];
    uint16_t h, l; split_bf(w, h, l);
    size_t p = bpos(RANK, k, n);
    ws[B1H + p] = h; ws[B1L + p] = l;
  }
  if (i < SZ_B2) {                        // V1^T: k=r, n=f
    int k = i >> 12, n = i & 4095;
    float w = cfc_V[(size_t)n * RANK + k];
    uint16_t h, l; split_bf(w, h, l);
    size_t p = bpos(D_FF, k, n);
    ws[B2H + p] = h; ws[B2L + p] = l;
  }
  if (i < SZ_B3) {                        // Us2: k=f, n=r
    int k = i >> 7, n = i & 127;
    float w = cproj_U[i] * cproj_S[n];
    uint16_t h, l; split_bf(w, h, l);
    size_t p = bpos(RANK, k, n);
    ws[B3H + p] = h; ws[B3L + p] = l;
  }
  if (i < SZ_B4) {                        // V2^T: k=r, n=d
    int k = i >> 10, n = i & 1023;
    float w = cproj_V[(size_t)n * RANK + k];
    uint16_t h, l; split_bf(w, h, l);
    size_t p = bpos(D_MODEL, k, n);
    ws[B4H + p] = h; ws[B4L + p] = l;
  }
}

// ---------------------------------------------------------------------------
// Fused low-rank MLP, all four GEMMs on the bf16 matrix pipe (bf16x3, fp32 acc):
//   t1 = x@Us1 ; h = gelu(t1@V1^T+b1) ; t2 = h@Us2 ; out = t2@V2^T+b2
// 32 rows per workgroup as 2 M-halves x 4 slices of 8 waves: both M-half waves
// share one WGP, so duplicate weight-fragment fetches hit the WGP$ and L2
// weight traffic per row is halved. h (32x4096) never touches HBM.
// ---------------------------------------------------------------------------
__global__ __launch_bounds__(NTHREADS) void fused_lowrank_mlp(
    const float* __restrict__ x,
    const float* __restrict__ cfc_bias,
    const float* __restrict__ cproj_bias,
    const uint16_t* __restrict__ ws,
    float* __restrict__ out) {

  __shared__ __align__(16) uint16_t xa_hi[2][2048],  xa_lo[2][2048];   // x A-frags   (16 KB)
  __shared__ __align__(16) uint16_t t1a_hi[2][2048], t1a_lo[2][2048];  // t1 A-frags  (16 KB)
  __shared__ __align__(16) uint16_t hs_hi[NWAVES][512], hs_lo[NWAVES][512]; // gelu(h) (16 KB)
  __shared__ __align__(16) uint16_t t2a_hi[2][2048], t2a_lo[2][2048];  // t2 A-frags  (16 KB)
  __shared__ float t2s[2][MTH * RANK];                                 // fp32 reduce (16 KB)

  const int tid    = threadIdx.x;
  const int wave   = tid >> 5;
  const int mhalf  = wave >> 2;           // which 16-row half of the 32-row tile
  const int wslice = wave & 3;            // N/F slice within the half
  const int lane   = tid & 31;
  const int l16    = lane & 15;
  const int mbase  = (lane >> 4) << 3;    // C/D row base for this half-wave
  const int row0   = blockIdx.x * MT;

  // ---------------- Phase 1: t1[32x128] = x_tile @ Us1 ----------------
  v8f acc0 = {}; v8f acc1 = {};
  const int n0a = wslice * 16, n0b = (wslice + 4) * 16;
  for (int kc = 0; kc < D_MODEL; kc += 128) {
    // stage x[row0..+31][kc..+127]: coalesced float4 loads, split hi/lo,
    // scatter directly into per-half A-fragment layout in LDS
    #pragma unroll
    for (int i = 0; i < 4; ++i) {
      int slot = tid + i * NTHREADS;          // 1024 float4 slots
      int m = slot >> 5;                      // 0..31
      int j = (slot & 31) << 2;               // 0..124
      float4 xv = *(const float4*)&x[(size_t)(row0 + m) * D_MODEL + kc + j];
      int mh = m >> 4, mloc = m & 15;
      int chunk = j >> 5, kin = j & 31;
      int lane_t = mloc | (((kin >> 3) & 1) << 4);
      int v = ((kin & 16) >> 2) + ((kin & 7) >> 1);
      int du = ((chunk * 32 + lane_t) * 8 + v) * 2;   // u16 index, 8B aligned
      uint16_t h0,l0,h1,l1,h2,l2,h3,l3;
      split_bf(xv.x, h0, l0); split_bf(xv.y, h1, l1);
      split_bf(xv.z, h2, l2); split_bf(xv.w, h3, l3);
      uint2 hv, lv;
      hv.x = (uint32_t)h0 | ((uint32_t)h1 << 16); hv.y = (uint32_t)h2 | ((uint32_t)h3 << 16);
      lv.x = (uint32_t)l0 | ((uint32_t)l1 << 16); lv.y = (uint32_t)l2 | ((uint32_t)l3 << 16);
      *(uint2*)&xa_hi[mh][du] = hv;
      *(uint2*)&xa_lo[mh][du] = lv;
    }
    __syncthreads();
    #pragma unroll
    for (int c = 0; c < 4; ++c) {
      int gchunk = (kc >> 5) + c;
      v16bf ah = load_frag(xa_hi[mhalf], c, lane);
      v16bf al = load_frag(xa_lo[mhalf], c, lane);
      v16bf b0h = load_frag(ws + B1H, gchunk * 8 + wslice, lane);
      v16bf b0l = load_frag(ws + B1L, gchunk * 8 + wslice, lane);
      v16bf b1h = load_frag(ws + B1H, gchunk * 8 + wslice + 4, lane);
      v16bf b1l = load_frag(ws + B1L, gchunk * 8 + wslice + 4, lane);
      acc0 = mma3(acc0, ah, al, b0h, b0l);
      acc1 = mma3(acc1, ah, al, b1h, b1l);
    }
    __syncthreads();
  }
  // park t1 (C layout -> A-fragment hi/lo planes), zero t2 reduction buffer
  #pragma unroll
  for (int v = 0; v < 8; ++v) {
    int kkA = n0a + l16, kkB = n0b + l16;
    store_a16(t1a_hi[mhalf], t1a_lo[mhalf], kkA >> 5, mbase + v, kkA & 31, acc0[v]);
    store_a16(t1a_hi[mhalf], t1a_lo[mhalf], kkB >> 5, mbase + v, kkB & 31, acc1[v]);
  }
  #pragma unroll
  for (int i = 0; i < (MT * RANK) / NTHREADS; ++i) ((float*)t2s)[tid + i * NTHREADS] = 0.0f;
  __syncthreads();

  // -- Phase 2: stream gelu(h) 16x32 per wave-iteration; K-split t2 partials --
  v8f t2a[8];
  #pragma unroll
  for (int t = 0; t < 8; ++t) t2a[t] = (v8f){};
  uint16_t* hwh = &hs_hi[wave][0];
  uint16_t* hwl = &hs_lo[wave][0];

  for (int f0 = wslice * 32; f0 < D_FF; f0 += 4 * 32) {
    // h[16 x 32] = t1 @ V1^T[:, f0:f0+32]
    v8f hA = {}; v8f hB = {};
    const int tA = f0 >> 4;
    #pragma unroll
    for (int c = 0; c < 4; ++c) {
      v16bf ah = load_frag(t1a_hi[mhalf], c, lane);
      v16bf al = load_frag(t1a_lo[mhalf], c, lane);
      v16bf b0h = load_frag(ws + B2H, c * (D_FF / 16) + tA, lane);
      v16bf b0l = load_frag(ws + B2L, c * (D_FF / 16) + tA, lane);
      v16bf b1h = load_frag(ws + B2H, c * (D_FF / 16) + tA + 1, lane);
      v16bf b1l = load_frag(ws + B2L, c * (D_FF / 16) + tA + 1, lane);
      hA = mma3(hA, ah, al, b0h, b0l);
      hB = mma3(hB, ah, al, b1h, b1l);
    }
    // bias + GELU (fast exact-erf), relayout C->A (hi/lo) via wave-private LDS
    const float bfA = cfc_bias[f0 + l16];
    const float bfB = cfc_bias[f0 + 16 + l16];
    #pragma unroll
    for (int v = 0; v < 8; ++v) {
      store_a16(hwh, hwl, 0, mbase + v, l16,      gelu_exact(hA[v] + bfA));
      store_a16(hwh, hwl, 0, mbase + v, 16 + l16, gelu_exact(hB[v] + bfB));
    }
    asm volatile("s_wait_dscnt 0x0" ::: "memory");   // cross-lane LDS visibility
    __builtin_amdgcn_wave_barrier();

    // t2 += gelu(h) @ Us2[f0:f0+32, :]
    v16bf ah2 = load_frag(hwh, 0, lane);
    v16bf al2 = load_frag(hwl, 0, lane);
    const int c3 = f0 >> 5;
    #pragma unroll
    for (int nt = 0; nt < 8; ++nt) {
      v16bf bh = load_frag(ws + B3H, c3 * 8 + nt, lane);
      v16bf bl = load_frag(ws + B3L, c3 * 8 + nt, lane);
      t2a[nt] = mma3(t2a[nt], ah2, al2, bh, bl);
    }
    __builtin_amdgcn_wave_barrier();                 // hw reads precede next overwrite
  }
  // reduce 4 per-half K-split partials (fp32) via LDS float atomics
  #pragma unroll
  for (int t = 0; t < 8; ++t)
    #pragma unroll
    for (int v = 0; v < 8; ++v)
      atomicAdd(&t2s[mhalf][(mbase + v) * RANK + t * 16 + l16], t2a[t][v]);
  __syncthreads();

  // convert reduced t2 -> A-fragment hi/lo planes (cooperative)
  #pragma unroll
  for (int i = 0; i < (MT * RANK) / NTHREADS; ++i) {
    int e = tid + i * NTHREADS;              // 0..4095
    int mh = e >> 11, rem = e & 2047;
    int m = rem >> 7, kk = rem & 127;
    store_a16(t2a_hi[mh], t2a_lo[mh], kk >> 5, m, kk & 31, t2s[mh][rem]);
  }
  __syncthreads();

  // ---------------- Phase 3: out = t2 @ V2^T + b2 ----------------
  for (int nt = wslice; nt < D_MODEL / 16; nt += 4) {
    v8f acc = {};
    #pragma unroll
    for (int c = 0; c < 4; ++c) {
      v16bf ah = load_frag(t2a_hi[mhalf], c, lane);
      v16bf al = load_frag(t2a_lo[mhalf], c, lane);
      v16bf bh = load_frag(ws + B4H, c * (D_MODEL / 16) + nt, lane);
      v16bf bl = load_frag(ws + B4L, c * (D_MODEL / 16) + nt, lane);
      acc = mma3(acc, ah, al, bh, bl);
    }
    const float bo = cproj_bias[nt * 16 + l16];
    #pragma unroll
    for (int v = 0; v < 8; ++v)
      out[(size_t)(row0 + mhalf * 16 + mbase + v) * D_MODEL + nt * 16 + l16] = acc[v] + bo;
  }
}

extern "C" void kernel_launch(void* const* d_in, const int* in_sizes, int n_in,
                              void* d_out, int out_size, void* d_ws, size_t ws_size,
                              hipStream_t stream) {
  const float* x          = (const float*)d_in[0];
  const float* cfc_U      = (const float*)d_in[1];
  const float* cfc_S      = (const float*)d_in[2];
  const float* cfc_V      = (const float*)d_in[3];
  const float* cfc_bias   = (const float*)d_in[4];
  const float* cproj_U    = (const float*)d_in[5];
  const float* cproj_S    = (const float*)d_in[6];
  const float* cproj_V    = (const float*)d_in[7];
  const float* cproj_bias = (const float*)d_in[8];
  float*     out  = (float*)d_out;
  uint16_t*  ws16 = (uint16_t*)d_ws;

  const int prep_elems = SZ_B2;   // largest segment element count (524288)
  prep_weights<<<(prep_elems + 255) / 256, 256, 0, stream>>>(
      cfc_U, cfc_S, cfc_V, cproj_U, cproj_S, cproj_V, ws16);

  fused_lowrank_mlp<<<M_TOTAL / MT, NTHREADS, 0, stream>>>(
      x, cfc_bias, cproj_bias, ws16, out);
}
